// SymbolicStateIntegrator_17875653886600
// MI455X (gfx1250) — matrix operand order repaired and checked by reading, not compiled, not run on hardware
//
#include <hip/hip_runtime.h>
#include <hip/hip_bf16.h>
#include <math.h>

#define BD 16      // batch
#define DD 1024    // feature dim
#define NN 4096    // nodes per batch

#define CHUNKS 16            // attention chunks per batch row
#define ROWS_PER_CHUNK (NN / CHUNKS)   // 256
#define PSTRIDE 1028         // floats per attention partial: m, l, pad, pad, acc[1024]

typedef __attribute__((ext_vector_type(16))) __bf16 v16bf;
typedef __attribute__((ext_vector_type(8)))  float  v8f;
typedef __attribute__((ext_vector_type(4)))  int    v4i;

#if defined(__gfx1250__) && __has_builtin(__builtin_amdgcn_global_load_async_to_lds_b128)
#define HAVE_ASYNC_LDS 1
#else
#define HAVE_ASYNC_LDS 0
#endif

__device__ __forceinline__ float gelu_exact(float x) {
    // matches jax.nn.gelu(approximate=False)
    return 0.5f * x * (1.0f + erff(x * 0.70710678118654752f));
}

__device__ __forceinline__ void wait_asynccnt_8() {
#if defined(__gfx1250__)
#if __has_builtin(__builtin_amdgcn_s_wait_asynccnt)
    __builtin_amdgcn_s_wait_asynccnt(8);
#else
    asm volatile("s_wait_asynccnt 0x8" ::: "memory");
#endif
#endif
}

__device__ __forceinline__ void wait_asynccnt_0() {
#if defined(__gfx1250__)
#if __has_builtin(__builtin_amdgcn_s_wait_asynccnt)
    __builtin_amdgcn_s_wait_asynccnt(0);
#else
    asm volatile("s_wait_asynccnt 0x0" ::: "memory");
#endif
#endif
}

#if HAVE_ASYNC_LDS
// Copy one 4 KB node row (1024 f32) global -> LDS with 8 async b128 transfers.
// Lane l moves bytes [l*16 + 512*j, +16); the 24-bit inst offset is added to
// BOTH the global and LDS address per the ISA, so one pointer pair suffices.
// Builtin signature (from compiler diagnostic): param 1 = AS(1) v4i* (global),
// param 2 = AS(3) v4i* (LDS), then i32 offset, i32 cpol.
__device__ __forceinline__ void async_copy_row(const float* gsrc,
                                               float* lds_dst, int lane) {
    typedef __attribute__((address_space(1))) v4i* gp_t;
    typedef __attribute__((address_space(3))) v4i* lp_t;
    gp_t g = (gp_t)(const_cast<float*>(gsrc) + lane * 4);
    lp_t l = (lp_t)(lds_dst + lane * 4);
    __builtin_amdgcn_global_load_async_to_lds_b128(g, l, 0,    0);
    __builtin_amdgcn_global_load_async_to_lds_b128(g, l, 512,  0);
    __builtin_amdgcn_global_load_async_to_lds_b128(g, l, 1024, 0);
    __builtin_amdgcn_global_load_async_to_lds_b128(g, l, 1536, 0);
    __builtin_amdgcn_global_load_async_to_lds_b128(g, l, 2048, 0);
    __builtin_amdgcn_global_load_async_to_lds_b128(g, l, 2560, 0);
    __builtin_amdgcn_global_load_async_to_lds_b128(g, l, 3072, 0);
    __builtin_amdgcn_global_load_async_to_lds_b128(g, l, 3584, 0);
}
#endif

// ---------------------------------------------------------------------------
// WMMA GEMM: C_part = A[16,K] x W[K,N]  (bf16 inputs, f32 accumulate)
// Block = 128 threads = 4 waves; wave w owns cols [blockIdx.x*64 + 16w, +16).
// K split across blockIdx.y into deterministic partial tiles in `part`.
// ---------------------------------------------------------------------------
__global__ void gemm16_bf16wmma_partial(const float* __restrict__ A,
                                        const float* __restrict__ W,
                                        float* __restrict__ part,
                                        int K, int N, int kchunk) {
    const int wave = threadIdx.x >> 5;
    const int lane = threadIdx.x & 31;
    const int col0 = blockIdx.x * 64 + wave * 16;
    const int m    = lane & 15;        // row (A) / col-in-tile (B,C)
    const int half = lane >> 4;        // half-wave selector
    const int kbeg = blockIdx.y * kchunk;
    const int kend = kbeg + kchunk;

    v8f c = {};
    for (int k0 = kbeg; k0 < kend; k0 += 32) {
        // A fragment (16x32 bf16): lanes 0-15 hold K 0..15, lanes 16-31 K 16..31
        v16bf a;
        const float* ap0 = A + (size_t)m * K + k0 + 16 * half;
        const float* ap1 = ap0 + 8;
#pragma unroll
        for (int i = 0; i < 8; ++i) a[i]     = (__bf16)ap0[i];
#pragma unroll
        for (int i = 0; i < 8; ++i) a[8 + i] = (__bf16)ap1[i];

        // B fragment (32x16 bf16): lane = column, half-wave selects K 0-15 / 16-31
        v16bf bm;
        const float* wp = W + (size_t)(k0 + 16 * half) * N + col0 + m;
#pragma unroll
        for (int j = 0; j < 16; ++j) bm[j] = (__bf16)wp[(size_t)j * N];

        if (k0 + 32 < kend)
            __builtin_prefetch(W + (size_t)(k0 + 32 + 16 * half) * N + col0 + m, 0, 1);

        c = __builtin_amdgcn_wmma_f32_16x16x32_bf16(false, a, false, bm,
                                                    (short)0, c, false, false);
    }

    // C layout: element r -> row r (lanes 0-15) / row r+8 (lanes 16-31), col = lane&15
    float* dst = part + ((size_t)blockIdx.y * 16 + 8 * half) * N + col0 + m;
#pragma unroll
    for (int r = 0; r < 8; ++r) dst[(size_t)r * N] = c[r];
}

// sum K-slices + bias (+ exact GELU) -> out[16,N]
__global__ void gemm16_epilogue(const float* __restrict__ part,
                                const float* __restrict__ bias,
                                float* __restrict__ out,
                                int N, int nslices, int do_gelu) {
    int idx = blockIdx.x * blockDim.x + threadIdx.x;
    if (idx >= 16 * N) return;
    int row = idx / N, col = idx - row * N;
    float s = 0.f;
    for (int k = 0; k < nslices; ++k) s += part[((size_t)k * 16 + row) * N + col];
    s += bias[col];
    if (do_gelu) s = gelu_exact(s);
    out[idx] = s;
}

// ---------------------------------------------------------------------------
// Fused attention pass 1: online-softmax partials, one HBM pass over nodes.
// Block = 256 threads = 8 waves; each wave owns 32 node rows of its chunk.
// On CDNA5 each wave double-buffers its current node row in LDS via
// GLOBAL_LOAD_ASYNC_TO_LDS_B128 (ASYNCcnt), issuing row t+1 before
// processing row t so the fetch overlaps the shuffle-reduce/exp chain.
// ---------------------------------------------------------------------------
__global__ void attn_partial(const float* __restrict__ nodes,
                             const float* __restrict__ query,
                             const int* __restrict__ counts,
                             float* __restrict__ part) {
    __shared__ float4 s_acc[8][256];   // 32 KB block-combine buffer
    __shared__ float  s_m[8], s_l[8];
#if HAVE_ASYNC_LDS
    __shared__ float  s_rows[8][2][DD];  // 64 KB: per-wave double-buffered row
#endif

    const int b     = blockIdx.x / CHUNKS;
    const int chunk = blockIdx.x % CHUNKS;
    const int wave  = threadIdx.x >> 5;
    const int lane  = threadIdx.x & 31;
    const int count = counts[b];

    // lane owns dims d = lane*4 + 128*j (coalesced float4 across the wave)
    const float4* q4 = (const float4*)(query + (size_t)b * DD);
    float4 qv[8];
#pragma unroll
    for (int j = 0; j < 8; ++j) qv[j] = q4[lane + 32 * j];

    float m = -INFINITY, l = 0.f;
    float4 acc[8];
#pragma unroll
    for (int j = 0; j < 8; ++j) acc[j] = make_float4(0.f, 0.f, 0.f, 0.f);

    const int n0     = chunk * ROWS_PER_CHUNK + wave * (ROWS_PER_CHUNK / 8);
    const int nseg   = n0 + ROWS_PER_CHUNK / 8;
    const int nvalid = (count < nseg) ? count : nseg;   // validity is a prefix

#if HAVE_ASYNC_LDS
    if (n0 < nvalid)
        async_copy_row(nodes + ((size_t)b * NN + n0) * DD, &s_rows[wave][0][0], lane);
#endif

    for (int t = 0; n0 + t < nvalid; ++t) {
        const int n = n0 + t;
#if HAVE_ASYNC_LDS
        const bool more = (n + 1 < nvalid);
        if (more)
            async_copy_row(nodes + ((size_t)b * NN + n + 1) * DD,
                           &s_rows[wave][(t + 1) & 1][0], lane);
        // async loads complete in order: <=8 outstanding => row t is in LDS
        if (more) wait_asynccnt_8();
        else      wait_asynccnt_0();
        const float4* src4 = (const float4*)(&s_rows[wave][t & 1][0]);
#else
        const float4* src4 = (const float4*)(nodes + ((size_t)b * NN + n) * DD);
#endif
        float4 rv[8];
        float dot = 0.f;
#pragma unroll
        for (int j = 0; j < 8; ++j) {
            float4 r4 = src4[lane + 32 * j];
            rv[j] = r4;
            dot += qv[j].x * r4.x + qv[j].y * r4.y + qv[j].z * r4.z + qv[j].w * r4.w;
        }
#pragma unroll
        for (int off = 16; off > 0; off >>= 1)
            dot += __shfl_xor(dot, off, 32);

        float s     = dot * 0.03125f;            // / sqrt(1024)
        float m_new = fmaxf(m, s);
        float corr  = __expf(m - m_new);         // m==-inf -> 0
        float w     = __expf(s - m_new);
        l = l * corr + w;
#pragma unroll
        for (int j = 0; j < 8; ++j) {
            acc[j].x = acc[j].x * corr + w * rv[j].x;
            acc[j].y = acc[j].y * corr + w * rv[j].y;
            acc[j].z = acc[j].z * corr + w * rv[j].z;
            acc[j].w = acc[j].w * corr + w * rv[j].w;
        }
        m = m_new;
    }

    if (lane == 0) s_m[wave] = m;
    __syncthreads();
    float m_blk = s_m[0];
#pragma unroll
    for (int w2 = 1; w2 < 8; ++w2) m_blk = fmaxf(m_blk, s_m[w2]);

    float scale = (m == -INFINITY) ? 0.f : __expf(m - m_blk);  // avoid inf-inf NaN
    if (lane == 0) s_l[wave] = l * scale;
#pragma unroll
    for (int j = 0; j < 8; ++j) {
        float4 v = acc[j];
        v.x *= scale; v.y *= scale; v.z *= scale; v.w *= scale;
        s_acc[wave][lane + 32 * j] = v;
    }
    __syncthreads();

    const int tid = threadIdx.x;
    float4 v = s_acc[0][tid];
#pragma unroll
    for (int w2 = 1; w2 < 8; ++w2) {
        float4 u = s_acc[w2][tid];
        v.x += u.x; v.y += u.y; v.z += u.z; v.w += u.w;
    }
    float l_blk = 0.f;
#pragma unroll
    for (int w2 = 0; w2 < 8; ++w2) l_blk += s_l[w2];

    float* base = part + (size_t)(b * CHUNKS + chunk) * PSTRIDE;
    if (tid == 0) { base[0] = m_blk; base[1] = l_blk; }
    ((float4*)(base + 4))[tid] = v;   // base multiple of 4 floats -> 16B aligned
}

// pass 2: combine 16 chunk-partials per batch row; readout + anchor
__global__ void attn_reduce(const float* __restrict__ part,
                            float* __restrict__ readout,
                            float* __restrict__ anchor) {
    const int b = blockIdx.x;
    const int tid = threadIdx.x;  // 256 threads, float4 per thread
    float m_g = -INFINITY;
    for (int c = 0; c < CHUNKS; ++c)
        m_g = fmaxf(m_g, part[(size_t)(b * CHUNKS + c) * PSTRIDE]);
    float l_g = 0.f;
    float4 v = make_float4(0.f, 0.f, 0.f, 0.f);
    for (int c = 0; c < CHUNKS; ++c) {
        const float* base = part + (size_t)(b * CHUNKS + c) * PSTRIDE;
        float mc = base[0];
        if (mc == -INFINITY) continue;
        float f = __expf(mc - m_g);
        l_g += base[1] * f;
        float4 u = ((const float4*)(base + 4))[tid];
        v.x += f * u.x; v.y += f * u.y; v.z += f * u.z; v.w += f * u.w;
    }
    float inv_l = 1.f / l_g;   // count >= 1 guaranteed
    ((float4*)(readout + (size_t)b * DD))[tid] =
        make_float4(v.x * inv_l, v.y * inv_l, v.z * inv_l, v.w * inv_l);
    if (tid == 0) anchor[b] = inv_l;   // max attn = exp(m_g - m_g)/l_g
}

// ---------------------------------------------------------------------------
// concat builders
// ---------------------------------------------------------------------------
__global__ void build_qin(const float* __restrict__ pooled,
                          const float* __restrict__ ps,
                          float* __restrict__ out) {
    int idx = blockIdx.x * blockDim.x + threadIdx.x;
    if (idx >= BD * 4 * DD) return;
    int b = idx >> 12, j = idx & 4095, seg = j >> 10, d = j & 1023;
    float p = pooled[b * DD + d], q = ps[b * DD + d];
    out[idx] = (seg == 0) ? p : (seg == 1) ? q : (seg == 2) ? (p - q) : (p + q);
}

__global__ void build_gsin(const float* __restrict__ readout,
                           const float* __restrict__ pooled,
                           float* __restrict__ out) {
    int idx = blockIdx.x * blockDim.x + threadIdx.x;
    if (idx >= BD * 4 * DD) return;
    int b = idx >> 12, j = idx & 4095, seg = j >> 10, d = j & 1023;
    float r = readout[b * DD + d], p = pooled[b * DD + d];
    out[idx] = (seg == 0) ? r : (seg == 1) ? p : (seg == 2) ? (r - p) : (p - r);
}

__global__ void build_ppin(const float* __restrict__ gcs,
                           const float* __restrict__ readout,
                           const float* __restrict__ ps,
                           float* __restrict__ out) {
    int idx = blockIdx.x * blockDim.x + threadIdx.x;
    if (idx >= BD * 3 * DD) return;
    int b = idx / (3 * DD), j = idx % (3 * DD), seg = j / DD, d = j % DD;
    out[idx] = (seg == 0) ? gcs[b * DD + d]
             : (seg == 1) ? readout[b * DD + d]
                          : ps[b * DD + d];
}

// ---------------------------------------------------------------------------
extern "C" void kernel_launch(void* const* d_in, const int* in_sizes, int n_in,
                              void* d_out, int out_size, void* d_ws, size_t ws_size,
                              hipStream_t stream) {
    const float* program_state = (const float*)d_in[1];
    const float* pooled        = (const float*)d_in[2];
    const float* nodes         = (const float*)d_in[3];
    const int*   counts        = (const int*)d_in[4];
    const float* gq_W1 = (const float*)d_in[5];
    const float* gq_b1 = (const float*)d_in[6];
    const float* gq_W2 = (const float*)d_in[7];
    const float* gq_b2 = (const float*)d_in[8];
    const float* gs_W1 = (const float*)d_in[9];
    const float* gs_b1 = (const float*)d_in[10];
    const float* gs_W2 = (const float*)d_in[11];
    const float* gs_b2 = (const float*)d_in[12];
    const float* pp_W1 = (const float*)d_in[13];
    const float* pp_b1 = (const float*)d_in[14];
    const float* pp_W2 = (const float*)d_in[15];
    const float* pp_b2 = (const float*)d_in[16];

    float* out      = (float*)d_out;
    float* gcs_out  = out;                 // [16,1024]
    float* readout  = out + BD * DD;       // [16,1024]
    float* anchor   = out + 2 * BD * DD;   // [16]

    // workspace layout (~2.0 MB of floats)
    float* ws     = (float*)d_ws;
    float* buf_in = ws;                        // 16*4096
    float* hbuf   = buf_in + BD * 4 * DD;      // 16*1024
    float* query  = hbuf + BD * DD;            // 16*1024
    float* gcs1   = query + BD * DD;           // 16*1024
    float* part_g = gcs1 + BD * DD;            // 8*16*1024 GEMM K-slices
    float* part_a = part_g + 8 * BD * DD;      // 256*PSTRIDE attention partials

    const int KS = 8;                          // K-split (512 waves per layer)
    dim3 gblk(128);

    // ---- graph_query MLP: query = MLP(concat(...)) ----
    build_qin<<<(BD * 4 * DD) / 256, 256, 0, stream>>>(pooled, program_state, buf_in);
    gemm16_bf16wmma_partial<<<dim3(DD / 64, KS), gblk, 0, stream>>>(buf_in, gq_W1, part_g, 4 * DD, DD, (4 * DD) / KS);
    gemm16_epilogue<<<(BD * DD) / 256, 256, 0, stream>>>(part_g, gq_b1, hbuf, DD, KS, 1);
    gemm16_bf16wmma_partial<<<dim3(DD / 64, KS), gblk, 0, stream>>>(hbuf, gq_W2, part_g, DD, DD, DD / KS);
    gemm16_epilogue<<<(BD * DD) / 256, 256, 0, stream>>>(part_g, gq_b2, query, DD, KS, 0);

    // ---- fused masked-softmax attention over nodes (single HBM pass) ----
    attn_partial<<<BD * CHUNKS, 256, 0, stream>>>(nodes, query, counts, part_a);
    attn_reduce<<<BD, 256, 0, stream>>>(part_a, readout, anchor);

    // ---- graph_state_proj MLP ----
    build_gsin<<<(BD * 4 * DD) / 256, 256, 0, stream>>>(readout, pooled, buf_in);
    gemm16_bf16wmma_partial<<<dim3(DD / 64, KS), gblk, 0, stream>>>(buf_in, gs_W1, part_g, 4 * DD, DD, (4 * DD) / KS);
    gemm16_epilogue<<<(BD * DD) / 256, 256, 0, stream>>>(part_g, gs_b1, hbuf, DD, KS, 1);
    gemm16_bf16wmma_partial<<<dim3(DD / 64, KS), gblk, 0, stream>>>(hbuf, gs_W2, part_g, DD, DD, DD / KS);
    gemm16_epilogue<<<(BD * DD) / 256, 256, 0, stream>>>(part_g, gs_b2, gcs1, DD, KS, 0);

    // ---- program_state_proj MLP ----
    build_ppin<<<(BD * 3 * DD) / 256, 256, 0, stream>>>(gcs1, readout, program_state, buf_in);
    gemm16_bf16wmma_partial<<<dim3(DD / 64, KS), gblk, 0, stream>>>(buf_in, pp_W1, part_g, 3 * DD, DD, (3 * DD) / KS);
    gemm16_epilogue<<<(BD * DD) / 256, 256, 0, stream>>>(part_g, pp_b1, hbuf, DD, KS, 1);
    gemm16_bf16wmma_partial<<<dim3(DD / 64, KS), gblk, 0, stream>>>(hbuf, pp_W2, part_g, DD, DD, DD / KS);
    gemm16_epilogue<<<(BD * DD) / 256, 256, 0, stream>>>(part_g, pp_b2, gcs_out, DD, KS, 0);
}
